// AttentionPool_11175504904448
// MI455X (gfx1250) — matrix operand order repaired and checked by reading, not compiled, not run on hardware
//
#include <hip/hip_runtime.h>
#include <hip/hip_fp16.h>
#include <math.h>

typedef _Float16 v8h  __attribute__((ext_vector_type(8)));
typedef _Float16 v16h __attribute__((ext_vector_type(16)));
typedef float    v8f  __attribute__((ext_vector_type(8)));

#define DIM 256
#define HID 128
#define ROWS_PER_BLOCK 128   // 8 wave32s x 16 rows
#define NPB 128              // nodes per block in pool kernel

// ---------------------------------------------------------------------------
// Kernel 1: scores[i] = gelu(x[i,:] @ W1 + b1) @ W2 + b2 via f16 WMMA
// ---------------------------------------------------------------------------
__global__ __launch_bounds__(256)
void score_kernel(const float* __restrict__ x, const float* __restrict__ W1,
                  const float* __restrict__ b1, const float* __restrict__ W2,
                  const float* __restrict__ b2, float* __restrict__ scores, int N)
{
    // W1^T (HID x DIM) in f16, XOR-swizzled in 16B chunks to kill bank conflicts.
    __shared__ _Float16 ldsW[HID * DIM];   // exactly 64 KB

    const int t = threadIdx.x;

    // Cooperative stage: W1 is (DIM x HID) row-major; store transposed+swizzled.
    for (int idx = t; idx < HID * DIM; idx += 256) {
        int n = idx & (HID - 1);       // hidden unit
        int k = idx >> 7;              // input dim
        _Float16 v = (_Float16)W1[k * HID + n];
        int chunk = (k >> 3) ^ (n & 31);               // swizzle 8-half chunks
        ldsW[n * DIM + chunk * 8 + (k & 7)] = v;
    }
    __syncthreads();

    const int wave = t >> 5;
    const int lane = t & 31;
    const int l    = lane & 15;   // position within half-wave
    const int hi   = lane >> 4;   // which half-wave
    const int row  = blockIdx.x * ROWS_PER_BLOCK + wave * 16 + l;  // A row for this lane
    const bool rvalid = (row < N);
    const float* xrow = x + (size_t)(rvalid ? row : 0) * DIM;

    // A fragments for all 8 K-tiles, ISA 16-bit A layout:
    //   lanes 0-15 : K = {kb..kb+7, kb+16..kb+23}
    //   lanes 16-31: K = {kb+8..kb+15, kb+24..kb+31}
    v16h a[8];
#pragma unroll
    for (int kt = 0; kt < 8; ++kt) {
        const int kb = kt * 32;
#pragma unroll
        for (int j = 0; j < 8; ++j) {
            float f0 = rvalid ? xrow[kb + hi * 8 + j]      : 0.0f;
            float f1 = rvalid ? xrow[kb + 16 + hi * 8 + j] : 0.0f;
            a[kt][j]     = (_Float16)f0;
            a[kt][j + 8] = (_Float16)f1;
        }
    }

    float sacc[8];
#pragma unroll
    for (int r = 0; r < 8; ++r) sacc[r] = 0.0f;

#pragma unroll
    for (int nt = 0; nt < 8; ++nt) {
        const int n = nt * 16 + l;                 // hidden column for this lane
        const _Float16* wrow = ldsW + n * DIM;
        v8f c = {};
#pragma unroll
        for (int kt = 0; kt < 8; ++kt) {
            // B layout: lanes 0-15 hold K = {kb..kb+7, kb+16..kb+23}, lanes 16-31 K = {kb+8..kb+15, kb+24..kb+31}
            const int c0 = kt * 4 + hi;            // logical 8-half chunk index
            v8h blo = *(const v8h*)(wrow + (((c0    ) ^ (n & 31)) * 8));
            v8h bhi = *(const v8h*)(wrow + (((c0 + 2) ^ (n & 31)) * 8));
            v16h b;
#pragma unroll
            for (int j = 0; j < 8; ++j) { b[j] = blo[j]; b[j + 8] = bhi[j]; }
            c = __builtin_amdgcn_wmma_f32_16x16x32_f16(
                    false, a[kt], false, b, (short)0, c, false, false);
        }
        const float bn  = b1[n];
        const float w2n = W2[n];
#pragma unroll
        for (int r = 0; r < 8; ++r) {
            float hv = c[r] + bn;
            float g  = 0.5f * hv * (1.0f + erff(hv * 0.7071067811865475f)); // exact GELU
            sacc[r] += g * w2n;
        }
    }

    // Reduce over the 16 lanes of each half-wave (columns of the hidden layer).
#pragma unroll
    for (int r = 0; r < 8; ++r) {
        float v = sacc[r];
        v += __shfl_xor(v, 1, 32);
        v += __shfl_xor(v, 2, 32);
        v += __shfl_xor(v, 4, 32);
        v += __shfl_xor(v, 8, 32);
        sacc[r] = v;
    }
    if (l == 0) {
        const float bb = b2[0];
        const int base = blockIdx.x * ROWS_PER_BLOCK + wave * 16 + hi * 8;
#pragma unroll
        for (int r = 0; r < 8; ++r) {
            int node = base + r;                   // VGPR r -> row (hi ? r+8 : r)
            if (node < N) scores[node] = sacc[r] + bb;
        }
    }
}

// ---------------------------------------------------------------------------
// Monotonic float <-> u32 encoding so atomicMax(u32) == float max
// ---------------------------------------------------------------------------
__device__ inline unsigned enc_f32(float f) {
    unsigned u = __float_as_uint(f);
    return (u & 0x80000000u) ? ~u : (u | 0x80000000u);
}
__device__ inline float dec_f32(unsigned u) {
    return __uint_as_float((u & 0x80000000u) ? (u & 0x7fffffffu) : ~u);
}

__global__ void init_kernel(float* __restrict__ out, unsigned* __restrict__ enc,
                            float* __restrict__ ssum, int Bseg)
{
    int i = blockIdx.x * blockDim.x + threadIdx.x;
    if (i < Bseg * DIM) out[i] = 0.0f;
    if (i < Bseg) { enc[i] = 0u; ssum[i] = 0.0f; }
}

// ---------------------------------------------------------------------------
// Kernel 3: per-segment max (sorted batch -> wave segmented scan, few atomics)
// ---------------------------------------------------------------------------
__global__ void segmax_kernel(const float* __restrict__ scores,
                              const int* __restrict__ batch,
                              unsigned* __restrict__ enc, int N)
{
    int i = blockIdx.x * blockDim.x + threadIdx.x;
    int lane = threadIdx.x & 31;
    int seg = -1; unsigned v = 0u;
    if (i < N) { seg = batch[i]; v = enc_f32(scores[i]); }
#pragma unroll
    for (int d = 1; d < 32; d <<= 1) {
        unsigned ov = __shfl_up(v, d, 32);
        int      os = __shfl_up(seg, d, 32);
        if (lane >= d && os == seg && ov > v) v = ov;
    }
    int nseg = __shfl_down(seg, 1, 32);
    if (seg >= 0 && (lane == 31 || nseg != seg)) atomicMax(&enc[seg], v);
}

// ---------------------------------------------------------------------------
// Kernel 4: per-segment sum of exp(score - max)
// ---------------------------------------------------------------------------
__global__ void sumexp_kernel(const float* __restrict__ scores,
                              const int* __restrict__ batch,
                              const unsigned* __restrict__ enc,
                              float* __restrict__ ssum, int N)
{
    int i = blockIdx.x * blockDim.x + threadIdx.x;
    int lane = threadIdx.x & 31;
    int seg = -1; float v = 0.0f;
    if (i < N) { seg = batch[i]; v = expf(scores[i] - dec_f32(enc[seg])); }
#pragma unroll
    for (int d = 1; d < 32; d <<= 1) {
        float ov = __shfl_up(v, d, 32);
        int   os = __shfl_up(seg, d, 32);
        if (lane >= d && os == seg) v += ov;
    }
    int nseg = __shfl_down(seg, 1, 32);
    if (seg >= 0 && (lane == 31 || nseg != seg)) atomicAdd(&ssum[seg], v);
}

// ---------------------------------------------------------------------------
// Kernel 5: pooled[seg, :] += alpha_i * x[i, :]   (thread t owns dim t)
// ---------------------------------------------------------------------------
__global__ __launch_bounds__(256)
void pool_kernel(const float* __restrict__ x, const int* __restrict__ batch,
                 const float* __restrict__ scores, const unsigned* __restrict__ enc,
                 const float* __restrict__ ssum, float* __restrict__ out, int N)
{
    __shared__ float s_alpha[NPB];
    __shared__ int   s_seg[NPB];
    const int t  = threadIdx.x;
    const int i0 = blockIdx.x * NPB;

    if (t < NPB) {
        int i = i0 + t;
        if (i < N) {
            int seg = batch[i];
            s_seg[t]   = seg;
            float m    = dec_f32(enc[seg]);
            s_alpha[t] = expf(scores[i] - m) / (ssum[seg] + 1e-8f);
        } else { s_seg[t] = -1; s_alpha[t] = 0.0f; }
    }
    __syncthreads();

    float acc = 0.0f;
    int cur = s_seg[0];
    for (int j = 0; j < NPB; ++j) {
        int i = i0 + j;
        if (i >= N) break;
        int seg = s_seg[j];
        if (seg != cur) {
            if (cur >= 0) atomicAdd(&out[(size_t)cur * DIM + t], acc);
            acc = 0.0f; cur = seg;
        }
        acc += s_alpha[j] * x[(size_t)i * DIM + t];
    }
    if (cur >= 0) atomicAdd(&out[(size_t)cur * DIM + t], acc);
}

// ---------------------------------------------------------------------------
extern "C" void kernel_launch(void* const* d_in, const int* in_sizes, int n_in,
                              void* d_out, int out_size, void* d_ws, size_t ws_size,
                              hipStream_t stream)
{
    const float* x     = (const float*)d_in[0];
    const int*   batch = (const int*)d_in[1];
    const float* W1    = (const float*)d_in[2];
    const float* b1    = (const float*)d_in[3];
    const float* W2    = (const float*)d_in[4];
    const float* b2    = (const float*)d_in[5];
    float* out = (float*)d_out;

    const int N    = in_sizes[0] / DIM;   // 600000
    const int Bseg = out_size / DIM;      // 2048

    // workspace: scores[N] f32, enc[Bseg] u32, ssum[Bseg] f32
    float*    scores = (float*)d_ws;
    unsigned* enc    = (unsigned*)(scores + N);
    float*    ssum   = (float*)(enc + Bseg);

    const int nb_score = (N + ROWS_PER_BLOCK - 1) / ROWS_PER_BLOCK;
    score_kernel<<<nb_score, 256, 0, stream>>>(x, W1, b1, W2, b2, scores, N);

    const int tot = Bseg * DIM;
    init_kernel<<<(tot + 255) / 256, 256, 0, stream>>>(out, enc, ssum, Bseg);

    segmax_kernel<<<(N + 255) / 256, 256, 0, stream>>>(scores, batch, enc, N);
    sumexp_kernel<<<(N + 255) / 256, 256, 0, stream>>>(scores, batch, enc, ssum, N);

    pool_kernel<<<(N + NPB - 1) / NPB, 256, 0, stream>>>(x, batch, scores, enc, ssum, out, N);
}
